// LlamaDecoderLayerOutputContext_85985245266071
// MI455X (gfx1250) — matrix operand order repaired and checked by reading, not compiled, not run on hardware
//
#include <hip/hip_runtime.h>

// ---------------- Types / constants ----------------
typedef __attribute__((ext_vector_type(16))) __bf16 v16bf;
typedef __attribute__((ext_vector_type(8)))  __bf16 v8bf;
typedef __attribute__((ext_vector_type(4)))  __bf16 v4bf;
typedef __attribute__((ext_vector_type(8)))  float  v8f;
typedef __attribute__((ext_vector_type(4)))  unsigned int v4u;
typedef __attribute__((ext_vector_type(8)))  int    v8i;
typedef __attribute__((ext_vector_type(4)))  int    v4i;

static constexpr int Bb  = 2;
static constexpr int Ss  = 2048;
static constexpr int Hh  = 2048;
static constexpr int NH  = 16;
static constexpr int HD  = 128;
static constexpr int FF  = 8192;
static constexpr int Mrows = Bb * Ss;            // 4096

// ---------------- Helpers ----------------
__device__ inline v16bf load_frag(const __bf16* p, int second_off) {
  v8bf lo = *(const v8bf*)p;
  v8bf hi = *(const v8bf*)(p + second_off);
  v16bf f;
#pragma unroll
  for (int e = 0; e < 8; ++e) { f[e] = lo[e]; f[e + 8] = hi[e]; }
  return f;
}

// Issue a TDM 2D tile load: `rows` rows of 64 bytes each, row pitch = k_total bf16
// elements (= k_total/4 8-byte units). data_size = 8B. Destination LDS rows are
// contiguous (our LDS tiles have a 64-byte row pitch).
__device__ inline void tdm_load_tile(unsigned lds_addr, const __bf16* gsrc,
                                     unsigned rows, unsigned k_total) {
  const unsigned long long ga = (unsigned long long)(const void*)gsrc;
  const unsigned td0 = k_total >> 2;   // tensor dim0 / row stride, in 8-byte units
  v4u g0;
  g0[0] = 1u;                                        // count=1, user descriptor
  g0[1] = lds_addr;                                  // LDS byte address
  g0[2] = (unsigned)(ga & 0xFFFFFFFFu);              // global addr lo
  g0[3] = (unsigned)((ga >> 32) & 0x01FFFFFFu) | (2u << 30);  // addr hi | type=2
  v8i g1;
  g1[0] = (int)(3u << 16);                           // workgroup_mask=0, data_size=8B
  g1[1] = (int)((td0 & 0xFFFFu) << 16);              // tensor_dim0[15:0]
  g1[2] = (int)((td0 >> 16) | ((rows & 0xFFFFu) << 16));  // dim0 hi | tensor_dim1 lo
  g1[3] = (int)((rows >> 16) | (8u << 16));          // dim1 hi | tile_dim0=8 (64B)
  g1[4] = (int)(rows & 0xFFFFu);                     // tile_dim1=rows, tile_dim2=0
  g1[5] = (int)td0;                                  // tensor_dim0_stride lo32
  g1[6] = 0;                                         // stride hi | dim1_stride lo
  g1[7] = 0;
  v4i z4 = {0, 0, 0, 0};
#if __clang_major__ >= 23
  v8i z8 = {0, 0, 0, 0, 0, 0, 0, 0};
  __builtin_amdgcn_tensor_load_to_lds(g0, g1, z4, z4, z8, 0);
#else
  __builtin_amdgcn_tensor_load_to_lds(g0, g1, z4, z4, 0);
#endif
}

// ---------------- fp32 -> bf16 conversion ----------------
__global__ void cvt_bf16_kernel(const float* __restrict__ in, __bf16* __restrict__ out, size_t n4) {
  size_t i = (size_t)blockIdx.x * blockDim.x + threadIdx.x;
  if (i >= n4) return;
  const float4 f = ((const float4*)in)[i];
  v4bf o; o[0] = (__bf16)f.x; o[1] = (__bf16)f.y; o[2] = (__bf16)f.z; o[3] = (__bf16)f.w;
  ((v4bf*)out)[i] = o;
}

// ---------------- RMSNorm (fp32 in, bf16 out) ----------------
__global__ __launch_bounds__(256)
void rmsnorm_bf16_kernel(const float* __restrict__ x, const float* __restrict__ w,
                         __bf16* __restrict__ out, int H) {
  const int row = blockIdx.x;
  const float* xr = x + (size_t)row * H;
  __shared__ float red[256];
  float ss = 0.f;
  for (int i = threadIdx.x; i < H; i += 256) { float v = xr[i]; ss += v * v; }
  red[threadIdx.x] = ss; __syncthreads();
  for (int off = 128; off > 0; off >>= 1) {
    if ((int)threadIdx.x < off) red[threadIdx.x] += red[threadIdx.x + off];
    __syncthreads();
  }
  const float r = rsqrtf(red[0] / (float)H + 1e-6f);
  __bf16* orow = out + (size_t)row * H;
  for (int i = threadIdx.x; i < H; i += 256) orow[i] = (__bf16)(xr[i] * r * w[i]);
}

// ---------------- Generic bf16 WMMA GEMM: C[m,n] = sum_k A[m,k]*W[n,k] ----------------
// TDM (tensor_load_to_lds) double-buffered operand staging; wave 0 drives the DMA.
// MODE 0: outF[m,n] = acc + res[m,n] ; MODE 1: outB[m,n] = bf16(acc)
template <int MODE>
__global__ __launch_bounds__(256)
void gemm_bf16_wmma_kernel(const __bf16* __restrict__ A, const __bf16* __restrict__ W,
                           float* __restrict__ outF, __bf16* __restrict__ outB,
                           const float* __restrict__ res,
                           int M, int N, int K) {
  __shared__ __align__(16) __bf16 As[2][64 * 32];
  __shared__ __align__(16) __bf16 Ws[2][128 * 32];
  const int tid  = threadIdx.x;
  const int lane = tid & 31, wv = tid >> 5;
  const int wm = wv >> 2, wn = wv & 3;
  const int l16 = lane & 15, hi = lane >> 4;
  const int bm0 = blockIdx.y * 64, bn0 = blockIdx.x * 128;

  const unsigned ldsA[2] = { (unsigned)(size_t)&As[0][0], (unsigned)(size_t)&As[1][0] };
  const unsigned ldsW[2] = { (unsigned)(size_t)&Ws[0][0], (unsigned)(size_t)&Ws[1][0] };

  v8f acc[2][2] = {};

  if (wv == 0) {  // prologue: fetch tile 0 into buffer 0
    tdm_load_tile(ldsA[0], &A[(size_t)bm0 * K], 64, (unsigned)K);
    tdm_load_tile(ldsW[0], &W[(size_t)bn0 * K], 128, (unsigned)K);
  }

  for (int k0 = 0; k0 < K; k0 += 32) {
    const int buf = (k0 >> 5) & 1;
    if (wv == 0) __builtin_amdgcn_s_wait_tensorcnt(0);
    __syncthreads();                      // tile `buf` resident in LDS for all waves

    if (wv == 0 && (k0 + 32) < K) {       // prefetch next tile into the other buffer
      tdm_load_tile(ldsA[buf ^ 1], &A[(size_t)bm0 * K + k0 + 32], 64, (unsigned)K);
      tdm_load_tile(ldsW[buf ^ 1], &W[(size_t)bn0 * K + k0 + 32], 128, (unsigned)K);
    }

    v16bf af[2], bf_[2];
#pragma unroll
    for (int mi = 0; mi < 2; ++mi)
      af[mi] = load_frag(&As[buf][(wm * 32 + mi * 16 + l16) * 32 + hi * 8], 16);
#pragma unroll
    for (int ni = 0; ni < 2; ++ni)
      bf_[ni] = load_frag(&Ws[buf][(wn * 32 + ni * 16 + l16) * 32 + hi * 8], 16);

#pragma unroll
    for (int mi = 0; mi < 2; ++mi)
#pragma unroll
      for (int ni = 0; ni < 2; ++ni)
        acc[mi][ni] = __builtin_amdgcn_wmma_f32_16x16x32_bf16(
            false, af[mi], false, bf_[ni], (short)0, acc[mi][ni], false, false);
    __syncthreads();                      // everyone done reading `buf`
  }

#pragma unroll
  for (int mi = 0; mi < 2; ++mi)
#pragma unroll
    for (int ni = 0; ni < 2; ++ni) {
      const int mbase = bm0 + wm * 32 + mi * 16 + hi * 8;
      const int n = bn0 + wn * 32 + ni * 16 + l16;
#pragma unroll
      for (int r = 0; r < 8; ++r) {
        const size_t idx = (size_t)(mbase + r) * N + n;
        const float v = acc[mi][ni][r];
        if (MODE == 0) outF[idx] = v + res[idx];
        else           outB[idx] = (__bf16)v;
      }
    }
}

// ---------------- Fused gate/up MLP GEMM with QuadActivation ----------------
__global__ __launch_bounds__(256)
void mlp_gateup_wmma_kernel(const __bf16* __restrict__ X, const __bf16* __restrict__ Wg,
                            const __bf16* __restrict__ Wu, __bf16* __restrict__ Out,
                            int M, int N, int K) {
  __shared__ __align__(16) __bf16 As[64 * 32];
  __shared__ __align__(16) __bf16 Gs[128 * 32];
  __shared__ __align__(16) __bf16 Us[128 * 32];
  const int tid  = threadIdx.x;
  const int lane = tid & 31, wv = tid >> 5;
  const int wm = wv >> 2, wn = wv & 3;
  const int l16 = lane & 15, hi = lane >> 4;
  const int bm0 = blockIdx.y * 64, bn0 = blockIdx.x * 128;

  v8f ag[2][2] = {}, au[2][2] = {};

  for (int k0 = 0; k0 < K; k0 += 32) {
    {
      int r = tid >> 2, c = (tid & 3) * 8;
      *(v8bf*)&As[r * 32 + c] = *(const v8bf*)&X[(size_t)(bm0 + r) * K + k0 + c];
      *(v8bf*)&Gs[r * 32 + c] = *(const v8bf*)&Wg[(size_t)(bn0 + r) * K + k0 + c];
      *(v8bf*)&Gs[(r + 64) * 32 + c] = *(const v8bf*)&Wg[(size_t)(bn0 + r + 64) * K + k0 + c];
      *(v8bf*)&Us[r * 32 + c] = *(const v8bf*)&Wu[(size_t)(bn0 + r) * K + k0 + c];
      *(v8bf*)&Us[(r + 64) * 32 + c] = *(const v8bf*)&Wu[(size_t)(bn0 + r + 64) * K + k0 + c];
    }
    __syncthreads();

    v16bf af[2], gf[2], uf[2];
#pragma unroll
    for (int mi = 0; mi < 2; ++mi)
      af[mi] = load_frag(&As[(wm * 32 + mi * 16 + l16) * 32 + hi * 8], 16);
#pragma unroll
    for (int ni = 0; ni < 2; ++ni) {
      gf[ni] = load_frag(&Gs[(wn * 32 + ni * 16 + l16) * 32 + hi * 8], 16);
      uf[ni] = load_frag(&Us[(wn * 32 + ni * 16 + l16) * 32 + hi * 8], 16);
    }
#pragma unroll
    for (int mi = 0; mi < 2; ++mi)
#pragma unroll
      for (int ni = 0; ni < 2; ++ni) {
        ag[mi][ni] = __builtin_amdgcn_wmma_f32_16x16x32_bf16(
            false, af[mi], false, gf[ni], (short)0, ag[mi][ni], false, false);
        au[mi][ni] = __builtin_amdgcn_wmma_f32_16x16x32_bf16(
            false, af[mi], false, uf[ni], (short)0, au[mi][ni], false, false);
      }
    __syncthreads();
  }

#pragma unroll
  for (int mi = 0; mi < 2; ++mi)
#pragma unroll
    for (int ni = 0; ni < 2; ++ni) {
      const int mbase = bm0 + wm * 32 + mi * 16 + hi * 8;
      const int n = bn0 + wn * 32 + ni * 16 + l16;
#pragma unroll
      for (int r = 0; r < 8; ++r) {
        const float g = ag[mi][ni][r];
        const float u = au[mi][ni][r];
        const float q = 0.125f * g * g + 0.25f * g + 0.5f;   // QuadActivation
        Out[(size_t)(mbase + r) * N + n] = (__bf16)(q * u);
      }
    }
}

// ---------------- RoPE on Q and K (bf16 [B,S,H] -> bf16 [B*NH, S, HD]) ----------------
__global__ void rope_qk_kernel(const __bf16* __restrict__ q, const __bf16* __restrict__ k,
                               const int* __restrict__ pos,
                               __bf16* __restrict__ qr, __bf16* __restrict__ kr) {
  size_t t = (size_t)blockIdx.x * blockDim.x + threadIdx.x;   // over B*S*NH*64
  const int i = (int)(t & 63); t >>= 6;
  const int h = (int)(t % NH); t /= NH;
  const int s = (int)(t % Ss);
  const int b = (int)(t / Ss);
  const float p = (float)pos[(size_t)b * Ss + s];
  const float inv = __expf(-((float)(2 * i) / (float)HD) * 9.210340371976184f); // ln(1e4)
  float sn, cs; __sincosf(p * inv, &sn, &cs);
  const size_t src = ((size_t)b * Ss + s) * Hh + (size_t)h * HD;
  const size_t dst = (((size_t)b * NH + h) * Ss + s) * HD;
  {
    const float x0 = (float)q[src + i], x1 = (float)q[src + i + 64];
    qr[dst + i]      = (__bf16)(x0 * cs - x1 * sn);
    qr[dst + i + 64] = (__bf16)(x1 * cs + x0 * sn);
  }
  {
    const float x0 = (float)k[src + i], x1 = (float)k[src + i + 64];
    kr[dst + i]      = (__bf16)(x0 * cs - x1 * sn);
    kr[dst + i + 64] = (__bf16)(x1 * cs + x0 * sn);
  }
}

// ---------------- V transpose: [B,S,H] -> [B*NH, HD, S] ----------------
__global__ void vtrans_kernel(const __bf16* __restrict__ v, __bf16* __restrict__ vt) {
  size_t t = (size_t)blockIdx.x * blockDim.x + threadIdx.x;   // over B*NH*HD*S
  const int s = (int)(t % Ss); size_t r = t / Ss;
  const int d = (int)(r % HD); r /= HD;
  const int h = (int)(r % NH);
  const int b = (int)(r / NH);
  vt[t] = v[((size_t)b * Ss + s) * Hh + (size_t)h * HD + d];
}

// ---------------- Fused attention (QK^T -> softmax_2relu -> PV), WMMA ----------------
// Q,K: [B*NH, S, HD] bf16 ; Vt: [B*NH, HD, S] bf16 ; Out: [B,S,H] bf16
__global__ __launch_bounds__(256)
void attention_wmma_kernel(const __bf16* __restrict__ Q, const __bf16* __restrict__ Kh,
                           const __bf16* __restrict__ Vt, __bf16* __restrict__ Out) {
  const int bh = blockIdx.x;
  const int b = bh / NH, h = bh % NH;
  const int wv = threadIdx.x >> 5, lane = threadIdx.x & 31;
  const int l16 = lane & 15, hi = lane >> 4;
  const int qbase = blockIdx.y * 128 + wv * 16;

  const __bf16* q  = Q  + ((size_t)bh * Ss + qbase) * HD;
  const __bf16* kp = Kh + (size_t)bh * Ss * HD;
  const __bf16* vt = Vt + (size_t)bh * HD * Ss;

  __shared__ __align__(16) __bf16 Pl_all[8][16 * 32];   // wave-private P tiles
  __bf16* Pl = Pl_all[wv];

  v16bf qf[4];
#pragma unroll
  for (int j = 0; j < 4; ++j)
    qf[j] = load_frag(&q[(size_t)l16 * HD + j * 32 + hi * 8], 16);

  v8f  oacc[8] = {};
  float rs[8] = {};
  const float scale = 0.08838834764831845f;   // 1/sqrt(128)
  const int nend = qbase + 16;                // causal bound (exclusive)

  for (int n0 = 0; n0 < nend; n0 += 32) {
    v8f sc[2] = {};
#pragma unroll
    for (int ni = 0; ni < 2; ++ni)
#pragma unroll
      for (int j = 0; j < 4; ++j) {
        const int n = n0 + ni * 16 + l16;
        v16bf kf = load_frag(&kp[(size_t)n * HD + j * 32 + hi * 8], 16);
        sc[ni] = __builtin_amdgcn_wmma_f32_16x16x32_bf16(
            false, qf[j], false, kf, (short)0, sc[ni], false, false);
      }

    // scale + causal mask + relu; accumulate row sums; stage P tile in LDS
#pragma unroll
    for (int ni = 0; ni < 2; ++ni) {
      const int n = n0 + ni * 16 + l16;
#pragma unroll
      for (int r = 0; r < 8; ++r) {
        const int m = qbase + r + hi * 8;
        float v = sc[ni][r] * scale;
        v = (n <= m && v > 0.f) ? v : 0.f;
        rs[r] += v;
        Pl[(r + hi * 8) * 32 + ni * 16 + l16] = (__bf16)v;
      }
    }
    asm volatile("s_wait_dscnt 0" ::: "memory");

    // reload P in A-fragment layout, multiply by V^T rows
    v16bf pf = load_frag(&Pl[l16 * 32 + hi * 8], 16);
#pragma unroll
    for (int c = 0; c < 8; ++c) {
      const int d = c * 16 + l16;
      v16bf vf = load_frag(&vt[(size_t)d * Ss + n0 + hi * 8], 16);
      oacc[c] = __builtin_amdgcn_wmma_f32_16x16x32_bf16(
          false, pf, false, vf, (short)0, oacc[c], false, false);
    }
    asm volatile("s_wait_dscnt 0" ::: "memory");
  }

#pragma unroll
  for (int off = 1; off < 16; off <<= 1)
#pragma unroll
    for (int r = 0; r < 8; ++r) rs[r] += __shfl_xor(rs[r], off, 32);

#pragma unroll
  for (int c = 0; c < 8; ++c) {
    const int d = c * 16 + l16;
#pragma unroll
    for (int r = 0; r < 8; ++r) {
      const int m = qbase + r + hi * 8;
      const float v = oacc[c][r] / (rs[r] + 1e-12f);
      Out[((size_t)b * Ss + m) * Hh + (size_t)h * HD + d] = (__bf16)v;
    }
  }
}

// ---------------- Host launch ----------------
extern "C" void kernel_launch(void* const* d_in, const int* in_sizes, int n_in,
                              void* d_out, int out_size, void* d_ws, size_t ws_size,
                              hipStream_t stream) {
  const float* hidden = (const float*)d_in[0];
  const int*   pos    = (const int*)d_in[2];
  const float* wq = (const float*)d_in[3];
  const float* wk = (const float*)d_in[4];
  const float* wv = (const float*)d_in[5];
  const float* wo = (const float*)d_in[6];
  const float* wg = (const float*)d_in[7];
  const float* wu = (const float*)d_in[8];
  const float* wd = (const float*)d_in[9];
  const float* ln1 = (const float*)d_in[10];
  const float* ln2 = (const float*)d_in[11];

  const size_t HH = (size_t)Hh * Hh;          // 4194304
  const size_t FH = (size_t)FF * Hh;          // 16777216
  const size_t MS = (size_t)Mrows * Hh;       // 8388608
  char* ws = (char*)d_ws;

  size_t off = 0;
  auto take = [&](size_t bytes) { size_t o = off; off += (bytes + 255) & ~(size_t)255; return o; };
  const size_t o_wq = take(HH * 2), o_wk = take(HH * 2), o_wv = take(HH * 2), o_wo = take(HH * 2);
  const size_t o_wg = take(FH * 2), o_wu = take(FH * 2), o_wd = take(FH * 2);
  const size_t o_x1 = take(MS * 2);
  const size_t o_q  = take(MS * 2), o_k = take(MS * 2), o_v = take(MS * 2);
  const size_t o_qr = take(MS * 2), o_kr = take(MS * 2), o_vt = take(MS * 2);
  const size_t o_h2 = take(MS * 4);
  // reuse dead buffers:
  const size_t o_attn = o_x1;   // x1 dead after QKV GEMMs
  const size_t o_x2   = o_q;    // q dead after RoPE
  const size_t o_hmid = o_k;    // k..vt span (5*MS*2 = 80MB) covers B*S*FF bf16 (64MB)

  __bf16* bwq = (__bf16*)(ws + o_wq); __bf16* bwk = (__bf16*)(ws + o_wk);
  __bf16* bwv = (__bf16*)(ws + o_wv); __bf16* bwo = (__bf16*)(ws + o_wo);
  __bf16* bwg = (__bf16*)(ws + o_wg); __bf16* bwu = (__bf16*)(ws + o_wu);
  __bf16* bwd = (__bf16*)(ws + o_wd);
  __bf16* x1  = (__bf16*)(ws + o_x1);
  __bf16* qb  = (__bf16*)(ws + o_q);  __bf16* kb = (__bf16*)(ws + o_k);  __bf16* vb = (__bf16*)(ws + o_v);
  __bf16* qr  = (__bf16*)(ws + o_qr); __bf16* kr = (__bf16*)(ws + o_kr); __bf16* vt = (__bf16*)(ws + o_vt);
  float*  h2  = (float*)(ws + o_h2);
  __bf16* attn = (__bf16*)(ws + o_attn);
  __bf16* x2   = (__bf16*)(ws + o_x2);
  __bf16* hmid = (__bf16*)(ws + o_hmid);

  auto cvt = [&](const float* src, __bf16* dst, size_t n) {
    cvt_bf16_kernel<<<(unsigned)((n / 4 + 255) / 256), 256, 0, stream>>>(src, dst, n / 4);
  };
  cvt(wq, bwq, HH); cvt(wk, bwk, HH); cvt(wv, bwv, HH); cvt(wo, bwo, HH);
  cvt(wg, bwg, FH); cvt(wu, bwu, FH); cvt(wd, bwd, FH);

  // x1 = rmsnorm(hidden) * ln1
  rmsnorm_bf16_kernel<<<Mrows, 256, 0, stream>>>(hidden, ln1, x1, Hh);

  // Q/K/V projections (bf16 out), TDM-staged WMMA GEMM
  const dim3 gProj(Hh / 128, Mrows / 64);
  gemm_bf16_wmma_kernel<1><<<gProj, 256, 0, stream>>>(x1, bwq, nullptr, qb, nullptr, Mrows, Hh, Hh);
  gemm_bf16_wmma_kernel<1><<<gProj, 256, 0, stream>>>(x1, bwk, nullptr, kb, nullptr, Mrows, Hh, Hh);
  gemm_bf16_wmma_kernel<1><<<gProj, 256, 0, stream>>>(x1, bwv, nullptr, vb, nullptr, Mrows, Hh, Hh);

  // RoPE + V transpose
  rope_qk_kernel<<<(Bb * Ss * NH * 64) / 256, 256, 0, stream>>>(qb, kb, pos, qr, kr);
  vtrans_kernel<<<(Bb * NH * HD * Ss) / 256, 256, 0, stream>>>(vb, vt);

  // Attention
  attention_wmma_kernel<<<dim3(Bb * NH, Ss / 128), 256, 0, stream>>>(qr, kr, vt, attn);

  // O projection + residual  -> h2 (f32)
  gemm_bf16_wmma_kernel<0><<<gProj, 256, 0, stream>>>(attn, bwo, h2, nullptr, hidden, Mrows, Hh, Hh);

  // x2 = rmsnorm(h2) * ln2
  rmsnorm_bf16_kernel<<<Mrows, 256, 0, stream>>>(h2, ln2, x2, Hh);

  // Fused gate/up + QuadActivation -> hmid (bf16 [M, FF])
  mlp_gateup_wmma_kernel<<<dim3(FF / 128, Mrows / 64), 256, 0, stream>>>(x2, bwg, bwu, hmid, Mrows, FF, Hh);

  // Down projection + residual -> d_out (f32)
  gemm_bf16_wmma_kernel<0><<<dim3(Hh / 128, Mrows / 64), 256, 0, stream>>>(
      hmid, bwd, (float*)d_out, nullptr, h2, Mrows, Hh, FF);
}
